// InternImageCustom_47416438948106
// MI455X (gfx1250) — compile-verified, matrix operand order checked by reading, not run on hardware
//
#include <hip/hip_runtime.h>
#include <hip/hip_bf16.h>
#include <math.h>

typedef unsigned short u16;
typedef __attribute__((ext_vector_type(16))) __bf16 v16bf;
typedef __attribute__((ext_vector_type(8)))  float  v8f;
typedef __attribute__((ext_vector_type(4)))  int    v4i;

#define NB   4
#define HH   56
#define WW   56
#define CC   192
#define NTOK (NB * HH * WW)   // 12544, multiple of 16
#define GG   12
#define CGC  16
#define KPTS 8
#define HIDD 768

// ---------- small device helpers ----------
__device__ __forceinline__ u16 f2bf(float f) {
    unsigned u = __float_as_uint(f);
    unsigned r = u + 0x7FFFu + ((u >> 16) & 1u);   // round to nearest even
    return (u16)(r >> 16);
}
__device__ __forceinline__ float bf2f(u16 u) {
    return __uint_as_float(((unsigned)u) << 16);
}
__device__ __forceinline__ float wred(float v) {
    #pragma unroll
    for (int o = 16; o > 0; o >>= 1) v += __shfl_xor(v, o);
    return v;
}
__device__ __forceinline__ float gelu_exact(float x) {
    return 0.5f * x * (1.0f + erff(x * 0.70710678118654752f));
}

union FragU { v16bf bf; v4i i2[2]; };

// ---------- weight transpose + fp32->bf16 : Wt[n*K+k] = W[k*N+n] ----------
__global__ void convert_wT_kernel(const float* __restrict__ W, u16* __restrict__ Wt,
                                  int Kk, int Nn) {
    int i = blockIdx.x * blockDim.x + threadIdx.x;
    if (i >= Kk * Nn) return;
    int n = i / Kk, k = i % Kk;
    Wt[(size_t)n * Kk + k] = f2bf(W[(size_t)k * Nn + n]);
}

// ---------- generic bf16 WMMA GEMM: D[M,N] = A[M,K] * Wt[N,K]^T + bias ----------
// act: 0 = none, 1 = exact GELU.  outF / outB nullable.
__global__ void gemm_bf16_kernel(const u16* __restrict__ A, const u16* __restrict__ Wt,
                                 const float* __restrict__ bias,
                                 float* __restrict__ outF, u16* __restrict__ outB,
                                 int M, int N, int K, int act) {
    int wave = threadIdx.x >> 5;
    int lane = threadIdx.x & 31;
    int t = blockIdx.x * 8 + wave;
    int ntn = N >> 4;
    int total = (M >> 4) * ntn;
    if (t >= total) return;                 // uniform per wave
    int tm = t / ntn, tn = t % ntn;
    int m0 = tm << 4, n0 = tn << 4;

    int half = lane & 15;                   // M (for A) / N (for B) within tile
    int kb   = (lane >> 4) << 3;            // 0 or 8 per ISA 16-bit fragment layout

    const u16* arow = A  + (size_t)(m0 + half) * K;
    const u16* brow = Wt + (size_t)(n0 + half) * K;

    v8f c = {};
    for (int k0 = 0; k0 < K; k0 += 32) {
        FragU a, b;
        const v4i* ap = (const v4i*)(arow + k0 + kb);   // K = kb..kb+7
        const v4i* bp = (const v4i*)(brow + k0 + kb);
        a.i2[0] = ap[0]; a.i2[1] = ap[2];               // +32B => K = kb+16..kb+23
        b.i2[0] = bp[0]; b.i2[1] = bp[2];
        c = __builtin_amdgcn_wmma_f32_16x16x32_bf16(
                false, a.bf, false, b.bf, (short)0, c, false, false);
    }

    int col   = n0 + half;
    int rbase = m0 + ((lane >> 4) << 3);    // C/D layout: rows 0-7 / 8-15
    float bv = bias ? bias[col] : 0.0f;
    #pragma unroll
    for (int r = 0; r < 8; ++r) {
        float v = c[r] + bv;
        if (act == 1) v = gelu_exact(v);
        size_t idx = (size_t)(rbase + r) * N + col;
        if (outF) outF[idx] = v;
        if (outB) outB[idx] = f2bf(v);
    }
}

// ---------- wave-per-token LayerNorm (C=192, 6 ch/lane), optional GELU ----------
__global__ void ln_kernel(const float* __restrict__ x,
                          const float* __restrict__ g, const float* __restrict__ b,
                          float eps, int act,
                          float* __restrict__ outF, u16* __restrict__ outB, int Ntok) {
    int wave = threadIdx.x >> 5, lane = threadIdx.x & 31;
    int n = blockIdx.x * (blockDim.x >> 5) + wave;
    if (n >= Ntok) return;
    size_t base = (size_t)n * CC;
    float v[6], s = 0.f, s2 = 0.f;
    #pragma unroll
    for (int i = 0; i < 6; ++i) {
        float t = x[base + lane + 32 * i];
        v[i] = t; s += t; s2 += t * t;
    }
    s = wred(s); s2 = wred(s2);
    float mu  = s * (1.0f / CC);
    float var = s2 * (1.0f / CC) - mu * mu;
    float rs  = rsqrtf(var + eps);
    #pragma unroll
    for (int i = 0; i < 6; ++i) {
        int c = lane + 32 * i;
        float t = (v[i] - mu) * rs * g[c] + b[c];
        if (act == 1) t = gelu_exact(t);
        if (outF) outF[base + c] = t;
        if (outB) outB[base + c] = f2bf(t);
    }
}

// ---------- residual post-norm: sum = xres + LN(d)*w1+b1 ; optional ln2(sum) -> bf16 ----------
__global__ void res_ln_kernel(const float* __restrict__ d, const float* __restrict__ xres,
                              const float* __restrict__ w1, const float* __restrict__ b1,
                              const float* __restrict__ w2, const float* __restrict__ b2,
                              float* __restrict__ sumOutF, u16* __restrict__ ln2OutB,
                              int Ntok) {
    int wave = threadIdx.x >> 5, lane = threadIdx.x & 31;
    int n = blockIdx.x * (blockDim.x >> 5) + wave;
    if (n >= Ntok) return;
    size_t base = (size_t)n * CC;
    float v[6], s = 0.f, s2 = 0.f;
    #pragma unroll
    for (int i = 0; i < 6; ++i) {
        float t = d[base + lane + 32 * i];
        v[i] = t; s += t; s2 += t * t;
    }
    s = wred(s); s2 = wred(s2);
    float mu  = s * (1.0f / CC);
    float var = s2 * (1.0f / CC) - mu * mu;
    float rs  = rsqrtf(var + 1e-5f);
    float u[6], us = 0.f, us2 = 0.f;
    #pragma unroll
    for (int i = 0; i < 6; ++i) {
        int c = lane + 32 * i;
        float t = (v[i] - mu) * rs * w1[c] + b1[c] + xres[base + c];
        u[i] = t; us += t; us2 += t * t;
        sumOutF[base + c] = t;
    }
    if (w2) {
        us = wred(us); us2 = wred(us2);
        float mu2  = us * (1.0f / CC);
        float var2 = us2 * (1.0f / CC) - mu2 * mu2;
        float rs2  = rsqrtf(var2 + 1e-5f);
        #pragma unroll
        for (int i = 0; i < 6; ++i) {
            int c = lane + 32 * i;
            ln2OutB[base + c] = f2bf((u[i] - mu2) * rs2 * w2[c] + b2[c]);
        }
    }
}

// ---------- depthwise 3x3 'SAME' (NHWC, HWIO weights) ----------
__global__ void dwconv_kernel(const float* __restrict__ x, const float* __restrict__ wgt,
                              const float* __restrict__ bias, float* __restrict__ out,
                              int Ntok) {
    int idx = blockIdx.x * blockDim.x + threadIdx.x;
    if (idx >= Ntok * CC) return;
    int n = idx / CC, c = idx % CC;
    int b = n / (HH * WW), hw = n % (HH * WW);
    int h = hw / WW, w = hw % WW;
    float acc = bias[c];
    #pragma unroll
    for (int kh = 0; kh < 3; ++kh) {
        int ih = h + kh - 1;
        if ((unsigned)ih >= HH) continue;
        #pragma unroll
        for (int kw = 0; kw < 3; ++kw) {
            int iw = w + kw - 1;
            if ((unsigned)iw >= WW) continue;
            acc += x[(((size_t)(b * HH + ih) * WW + iw)) * CC + c] *
                   wgt[(kh * 3 + kw) * CC + c];
        }
    }
    out[(size_t)n * CC + c] = acc;
}

// ---------- cfs = sigmoid(x1 @ cfs_w^T + cfs_b), per (token, group) ----------
__global__ void cfs_kernel(const u16* __restrict__ x1b, const float* __restrict__ cw,
                           const float* __restrict__ cb, float* __restrict__ cfs,
                           int Ntok) {
    int idx = blockIdx.x * blockDim.x + threadIdx.x;
    if (idx >= Ntok * GG) return;
    int n = idx / GG, g = idx % GG;
    const u16* xr = x1b + (size_t)n * CC;
    const float* wr = cw + (size_t)g * CC;
    float acc = cb[g];
    for (int c = 0; c < CC; ++c) acc += bf2f(xr[c]) * wr[c];
    cfs[(size_t)n * GG + g] = 1.0f / (1.0f + expf(-acc));
}

// ---------- DCNv3 sampling: softmax(mask) * bilinear(xp) + center-feature-scale ----------
__global__ void sample_kernel(const float* __restrict__ xp, const float* __restrict__ off,
                              const float* __restrict__ mlin, const float* __restrict__ cfs,
                              u16* __restrict__ yb, int Ntok) {
    int idx = blockIdx.x * blockDim.x + threadIdx.x;
    if (idx >= Ntok * CC) return;
    int n = idx / CC, c = idx % CC;
    int g = c >> 4;                                   // CG = 16
    int b = n / (HH * WW), hw = n % (HH * WW);
    int h = hw / WW, w = hw % WW;

    // softmax over the 8 kernel points of this group
    const float* mr = mlin + (size_t)n * (GG * KPTS) + g * KPTS;
    float m[KPTS], mx = -3.0e38f;
    #pragma unroll
    for (int k = 0; k < KPTS; ++k) { m[k] = mr[k]; mx = fmaxf(mx, m[k]); }
    float se = 0.f;
    #pragma unroll
    for (int k = 0; k < KPTS; ++k) { m[k] = expf(m[k] - mx); se += m[k]; }
    float inv = 1.0f / se;

    const float* offr = off + (size_t)n * CC + g * (KPTS * 2);
    const float gx[KPTS] = {-1.f,-1.f,-1.f, 0.f, 0.f, 1.f, 1.f, 1.f};
    const float gy[KPTS] = {-1.f, 0.f, 1.f,-1.f, 1.f,-1.f, 0.f, 1.f};
    const size_t bbase = (size_t)b * (HH * WW);

    float acc = 0.f;
    #pragma unroll
    for (int k = 0; k < KPTS; ++k) {
        float px = (float)w + 1.0f + gx[k] + offr[2 * k + 0];   // padded coords
        float py = (float)h + 1.0f + gy[k] + offr[2 * k + 1];
        float x0 = floorf(px), y0 = floorf(py);
        float wx = px - x0, wy = py - y0;
        int ix0 = (int)x0, iy0 = (int)y0;
        float v00 = 0.f, v01 = 0.f, v10 = 0.f, v11 = 0.f;
        {   // fetch with zero outside [0,H)x[0,W) in unpadded coords (ring + OOB both 0)
            int ih = iy0 - 1, iw = ix0 - 1;
            if ((unsigned)ih < HH && (unsigned)iw < WW)
                v00 = xp[(bbase + (size_t)ih * WW + iw) * CC + c];
            if ((unsigned)ih < HH && (unsigned)(iw + 1) < WW)
                v01 = xp[(bbase + (size_t)ih * WW + iw + 1) * CC + c];
            if ((unsigned)(ih + 1) < HH && (unsigned)iw < WW)
                v10 = xp[(bbase + (size_t)(ih + 1) * WW + iw) * CC + c];
            if ((unsigned)(ih + 1) < HH && (unsigned)(iw + 1) < WW)
                v11 = xp[(bbase + (size_t)(ih + 1) * WW + iw + 1) * CC + c];
        }
        float bil = v00 * (1.f - wy) * (1.f - wx) + v01 * (1.f - wy) * wx
                  + v10 * wy * (1.f - wx)         + v11 * wy * wx;
        acc += m[k] * inv * bil;
    }
    float cf  = cfs[(size_t)n * GG + g];
    float xpv = xp[(size_t)n * CC + c];
    yb[(size_t)n * CC + c] = f2bf(acc * (1.f - cf) + xpv * cf);
}

// ---------- host orchestration ----------
extern "C" void kernel_launch(void* const* d_in, const int* in_sizes, int n_in,
                              void* d_out, int out_size, void* d_ws, size_t ws_size,
                              hipStream_t stream) {
    const float* x          = (const float*)d_in[0];
    const float* norm1_w    = (const float*)d_in[1];
    const float* norm1_b    = (const float*)d_in[2];
    const float* norm2_w    = (const float*)d_in[3];
    const float* norm2_b    = (const float*)d_in[4];
    const float* rpn1_w     = (const float*)d_in[5];
    const float* rpn1_b     = (const float*)d_in[6];
    const float* rpn2_w     = (const float*)d_in[7];
    const float* rpn2_b     = (const float*)d_in[8];
    const float* in_proj_w  = (const float*)d_in[9];
    const float* in_proj_b  = (const float*)d_in[10];
    const float* dw_w       = (const float*)d_in[11];
    const float* dw_b       = (const float*)d_in[12];
    const float* dw_ln_w    = (const float*)d_in[13];
    const float* dw_ln_b    = (const float*)d_in[14];
    const float* off_w      = (const float*)d_in[15];
    const float* off_b      = (const float*)d_in[16];
    const float* mask_w     = (const float*)d_in[17];
    const float* mask_b     = (const float*)d_in[18];
    const float* cfs_w      = (const float*)d_in[19];
    const float* cfs_b      = (const float*)d_in[20];
    const float* out_proj_w = (const float*)d_in[21];
    const float* out_proj_b = (const float*)d_in[22];
    const float* fc1_w      = (const float*)d_in[23];
    const float* fc1_b      = (const float*)d_in[24];
    const float* fc2_w      = (const float*)d_in[25];
    const float* fc2_b      = (const float*)d_in[26];

    char* ws = (char*)d_ws;
    size_t cur = 0;
    auto alloc = [&](size_t bytes) -> void* {
        void* p = ws + cur;
        cur = (cur + bytes + 255) & ~(size_t)255;
        return p;
    };

    float* xnF   = (float*)alloc((size_t)NTOK * CC * 4);
    u16*   xnB   = (u16*)  alloc((size_t)NTOK * CC * 2);
    float* xp    = (float*)alloc((size_t)NTOK * CC * 4);
    float* dwt   = (float*)alloc((size_t)NTOK * CC * 4);
    u16*   x1B   = (u16*)  alloc((size_t)NTOK * CC * 2);
    float* offB  = (float*)alloc((size_t)NTOK * CC * 4);           // G*K*2 == 192
    float* mlin  = (float*)alloc((size_t)NTOK * (GG * KPTS) * 4);  // 96
    float* cfsB  = (float*)alloc((size_t)NTOK * GG * 4);
    u16*   yB    = (u16*)  alloc((size_t)NTOK * CC * 2);
    float* dproj = (float*)alloc((size_t)NTOK * CC * 4);
    float* x2    = (float*)alloc((size_t)NTOK * CC * 4);
    u16*   x2nB  = (u16*)  alloc((size_t)NTOK * CC * 2);
    u16*   hB    = (u16*)  alloc((size_t)NTOK * HIDD * 2);
    float* dmlp  = (float*)alloc((size_t)NTOK * CC * 4);
    u16*   wInT  = (u16*)  alloc((size_t)CC * CC * 2);
    u16*   wOffT = (u16*)  alloc((size_t)CC * CC * 2);
    u16*   wMskT = (u16*)  alloc((size_t)(GG * KPTS) * CC * 2);
    u16*   wOutT = (u16*)  alloc((size_t)CC * CC * 2);
    u16*   wFc1T = (u16*)  alloc((size_t)HIDD * CC * 2);
    u16*   wFc2T = (u16*)  alloc((size_t)CC * HIDD * 2);

    const int THR = 256;
    auto cvt = [&](const float* Wsrc, u16* Wdst, int Kk, int Nn) {
        int tot = Kk * Nn;
        convert_wT_kernel<<<(tot + THR - 1) / THR, THR, 0, stream>>>(Wsrc, Wdst, Kk, Nn);
    };
    auto gemm = [&](const u16* A, const u16* Wt, const float* bias,
                    float* oF, u16* oB, int M, int N, int K, int act) {
        int tiles = (M / 16) * (N / 16);
        gemm_bf16_kernel<<<(tiles + 7) / 8, 256, 0, stream>>>(A, Wt, bias, oF, oB, M, N, K, act);
    };

    cvt(in_proj_w,  wInT,  CC,   CC);
    cvt(off_w,      wOffT, CC,   CC);
    cvt(mask_w,     wMskT, CC,   GG * KPTS);
    cvt(out_proj_w, wOutT, CC,   CC);
    cvt(fc1_w,      wFc1T, CC,   HIDD);
    cvt(fc2_w,      wFc2T, HIDD, CC);

    int lnBlocks = (NTOK + 3) / 4;           // 4 waves per 128-thread block, wave per token
    int ewBlocks = (NTOK * CC + THR - 1) / THR;

    // 1) xn = LN(x, norm1)  (fp32 for dwconv, bf16 for GEMM)
    ln_kernel<<<lnBlocks, 128, 0, stream>>>(x, norm1_w, norm1_b, 1e-5f, 0, xnF, xnB, NTOK);
    // 2) xp = xn @ in_proj + b
    gemm(xnB, wInT, in_proj_b, xp, nullptr, NTOK, CC, CC, 0);
    // 3) depthwise conv -> LN(1e-6) -> GELU  => x1 (bf16)
    dwconv_kernel<<<ewBlocks, THR, 0, stream>>>(xnF, dw_w, dw_b, dwt, NTOK);
    ln_kernel<<<lnBlocks, 128, 0, stream>>>(dwt, dw_ln_w, dw_ln_b, 1e-6f, 1, nullptr, x1B, NTOK);
    // 4) offset / mask / cfs heads
    gemm(x1B, wOffT, off_b,  offB, nullptr, NTOK, CC,        CC, 0);
    gemm(x1B, wMskT, mask_b, mlin, nullptr, NTOK, GG * KPTS, CC, 0);
    cfs_kernel<<<(NTOK * GG + THR - 1) / THR, THR, 0, stream>>>(x1B, cfs_w, cfs_b, cfsB, NTOK);
    // 5) deformable sampling + center feature scale => y (bf16)
    sample_kernel<<<ewBlocks, THR, 0, stream>>>(xp, offB, mlin, cfsB, yB, NTOK);
    // 6) out projection
    gemm(yB, wOutT, out_proj_b, dproj, nullptr, NTOK, CC, CC, 0);
    // 7) x2 = x + LN(dproj, rpn1);  x2n = LN(x2, norm2) (bf16)
    res_ln_kernel<<<lnBlocks, 128, 0, stream>>>(dproj, x, rpn1_w, rpn1_b,
                                                norm2_w, norm2_b, x2, x2nB, NTOK);
    // 8) MLP: h = gelu(x2n @ fc1 + b) (bf16); dmlp = h @ fc2 + b
    gemm(x2nB, wFc1T, fc1_b, nullptr, hB, NTOK, HIDD, CC, 1);
    gemm(hB,   wFc2T, fc2_b, dmlp, nullptr, NTOK, CC, HIDD, 0);
    // 9) out = x2 + LN(dmlp, rpn2)
    res_ln_kernel<<<lnBlocks, 128, 0, stream>>>(dmlp, x2, rpn2_w, rpn2_b,
                                                nullptr, nullptr, (float*)d_out, nullptr, NTOK);
}
